// LowRankAttention_17806934409917
// MI455X (gfx1250) — matrix operand order repaired and checked by reading, not compiled, not run on hardware
//
#include <hip/hip_runtime.h>
#include <hip/hip_bf16.h>
#include <math.h>

#define D_MODEL 1024
#define N_HEADS 16
#define RANK    16
#define D_K     64
#define BATCH   2
#define SEQ     2048
#define ROWS    (BATCH*SEQ)   // 4096

typedef __attribute__((ext_vector_type(16))) __bf16 bf16x16;
typedef __attribute__((ext_vector_type(8)))  float  f32x8;
typedef int i32x4 __attribute__((vector_size(16)));   // matches builtin param

union FragAB { bf16x16 v; unsigned int u[8]; };

// ---- CDNA5 async global->LDS path (guarded; manual copy fallback) ----------
#if defined(__HIP_DEVICE_COMPILE__) && \
    __has_builtin(__builtin_amdgcn_global_load_async_to_lds_b128)
#define HAVE_ASYNC 1
#else
#define HAVE_ASYNC 0
#endif

#if HAVE_ASYNC
#define AS1V4(p) ((__attribute__((address_space(1))) i32x4*)(unsigned long long)(p))
#define AS3V4(p) ((__attribute__((address_space(3))) i32x4*)(unsigned int)(unsigned long long)(p))
#endif

__device__ __forceinline__ void cp_b128(const unsigned short* g, unsigned short* l) {
#if HAVE_ASYNC
  __builtin_amdgcn_global_load_async_to_lds_b128(AS1V4(g), AS3V4(l), 0, 0);
#else
  *(uint4*)l = *(const uint4*)g;     // callers guarantee 16B alignment
#endif
}

template <int N> __device__ __forceinline__ void wait_async() {
#if HAVE_ASYNC
#if __has_builtin(__builtin_amdgcn_s_wait_asynccnt)
  __builtin_amdgcn_s_wait_asynccnt(N);
#else
  asm volatile("s_wait_asynccnt %0" :: "n"(N) : "memory");
#endif
#endif
}

__device__ __forceinline__ unsigned short f2bf(float f) {
  unsigned int u = __float_as_uint(f);
  u += 0x7FFFu + ((u >> 16) & 1u);   // round-to-nearest-even
  return (unsigned short)(u >> 16);
}
__device__ __forceinline__ unsigned int pk2(float a, float b) {
  return (unsigned int)f2bf(a) | ((unsigned int)f2bf(b) << 16);
}
__device__ __forceinline__ f32x8 zero8() {
  f32x8 z;
#pragma unroll
  for (int i = 0; i < 8; ++i) z[i] = 0.0f;
  return z;
}

// ---------------------------------------------------------------- kernel 1
__global__ __launch_bounds__(256) void k_convert_x(const float* __restrict__ x,
                                                   unsigned short* __restrict__ xb) {
  int idx = blockIdx.x * 256 + threadIdx.x;
  xb[idx] = f2bf(x[idx]);
}

// ---------------------------------------------------------------- kernel 2
// WfT rows 0..255: (WQ_h@WQr)^T, 256..511: (WK_h@WKr)^T, 512..1535: WV^T.
// WOt: WO^T. bQrw/bKrw: fused biases.
__global__ __launch_bounds__(256) void k_fuse_weights(
    const float* __restrict__ WQ, const float* __restrict__ WK,
    const float* __restrict__ WV, const float* __restrict__ WO,
    const float* __restrict__ WQr, const float* __restrict__ WKr,
    const float* __restrict__ bQ,  const float* __restrict__ bK,
    unsigned short* __restrict__ WfT, unsigned short* __restrict__ WOt,
    float* __restrict__ bQrw, float* __restrict__ bKrw) {
  int idx = blockIdx.x * 256 + threadIdx.x;   // (1536+1024)*1024 threads
  int c = idx >> 10, d = idx & 1023;
  if (c < 256) {
    int h = c >> 4, r = c & 15;
    float acc = 0.f;
#pragma unroll 8
    for (int k = 0; k < D_K; ++k) acc += WQ[d * 1024 + h * 64 + k] * WQr[k * 16 + r];
    WfT[c * 1024 + d] = f2bf(acc);
    if (d == 0) {
      float ba = 0.f;
      for (int k = 0; k < D_K; ++k) ba += bQ[h * 64 + k] * WQr[k * 16 + r];
      bQrw[c] = ba;
    }
  } else if (c < 512) {
    int c2 = c - 256, h = c2 >> 4, r = c2 & 15;
    float acc = 0.f;
#pragma unroll 8
    for (int k = 0; k < D_K; ++k) acc += WK[d * 1024 + h * 64 + k] * WKr[k * 16 + r];
    WfT[c * 1024 + d] = f2bf(acc);
    if (d == 0) {
      float ba = 0.f;
      for (int k = 0; k < D_K; ++k) ba += bK[h * 64 + k] * WKr[k * 16 + r];
      bKrw[c2] = ba;
    }
  } else if (c < 1536) {
    WfT[c * 1024 + d] = f2bf(WV[d * 1024 + (c - 512)]);
  } else {
    int c2 = c - 1536;
    WOt[c2 * 1024 + d] = f2bf(WO[d * 1024 + c2]);
  }
}

// ------------------------------------------------------- shared GEMM mainloop
// 256 threads = 8 waves. Block tile 128x64, wave tile 16x64 (4 wmma acc).
// Double-buffered LDS fed by async B128 copies. Stride 40 ushorts = 80B rows:
// 16B aligned for B128 and bank-conflict-free for 16-lane strided reads.
__device__ __forceinline__ void gemm_tile_128x64(
    const unsigned short* __restrict__ A, int lda, int rowBase,
    const unsigned short* __restrict__ Bt, int ldb, int colBase,
    int Kdim, f32x8* acc, unsigned short* sA, unsigned short* sB) {
  const int tid = threadIdx.x;
  const int lane = tid & 31, wave = tid >> 5;
  const int half = lane >> 4, l16 = lane & 15;
#pragma unroll
  for (int t = 0; t < 4; ++t) acc[t] = zero8();

  auto issueTile = [&](int kb, int buf) {
    unsigned short* dA = sA + buf * (128 * 40);
    unsigned short* dB = sB + buf * (64 * 40);
#pragma unroll
    for (int j = 0; j < 2; ++j) {                 // A tile: 128 x 32
      int e = (tid + 256 * j) * 8;
      int r = e >> 5, k = e & 31;
      cp_b128(A + (size_t)(rowBase + r) * lda + kb + k, dA + r * 40 + k);
    }
    {                                              // B tile: 64 x 32
      int e = tid * 8;
      int c = e >> 5, k = e & 31;
      cp_b128(Bt + (size_t)(colBase + c) * ldb + kb + k, dB + c * 40 + k);
    }
  };

  issueTile(0, 0);
  const int nIter = Kdim >> 5;
  for (int it = 0; it < nIter; ++it) {
    int buf = it & 1;
    __syncthreads();                // all waves done with compute of it-1
    bool more = (it + 1) < nIter;
    if (more) issueTile((it + 1) << 5, buf ^ 1);
    if (more) wait_async<3>(); else wait_async<0>();   // current tile resident
    __syncthreads();                // all waves' current tile visible
    const unsigned short* dA = sA + buf * (128 * 40);
    const unsigned short* dB = sB + buf * (64 * 40);
    FragAB a;                       // A frag: M = lane%16, contiguous k = 16*half + s
#pragma unroll
    for (int j = 0; j < 4; ++j) {
      int k0 = half * 16 + 2 * j;
      a.u[j]     = *(const unsigned int*)(dA + (wave * 16 + l16) * 40 + k0);
      a.u[j + 4] = *(const unsigned int*)(dA + (wave * 16 + l16) * 40 + 8 + k0);
    }
#pragma unroll
    for (int t = 0; t < 4; ++t) {
      FragAB b;
#pragma unroll
      for (int j = 0; j < 8; ++j) {
        int k0 = half * 16 + 2 * j;
        b.u[j] = *(const unsigned int*)(dB + (t * 16 + l16) * 40 + k0);
      }
      acc[t] = __builtin_amdgcn_wmma_f32_16x16x32_bf16(
          false, a.v, false, b.v, (short)0, acc[t], false, false);
    }
  }
}

// ---------------------------------------------------------------- kernel 3
// C = x @ [FQ|FK|WV]; store Qr (b,h,n,16), Krp rank-padded (b,h,n,32),
// and V transposed (b,h,v,n) as packed 16B rows for async B128 reload.
__global__ __launch_bounds__(256) void k_gemm_qkv(
    const unsigned short* __restrict__ xb, const unsigned short* __restrict__ WfT,
    const float* __restrict__ bQr, const float* __restrict__ bKr,
    const float* __restrict__ bV,
    unsigned short* __restrict__ Qr, unsigned short* __restrict__ Krp,
    unsigned short* __restrict__ VT) {
  __shared__ __attribute__((aligned(16))) unsigned short sA[2 * 128 * 40];
  __shared__ __attribute__((aligned(16))) unsigned short sB[2 * 64 * 40];
  int bm = blockIdx.x & 31;          // 4096/128
  int bn = blockIdx.x >> 5;          // 1536/64
  f32x8 acc[4];
  gemm_tile_128x64(xb, 1024, bm * 128, WfT, 1024, bn * 64, 1024, acc, sA, sB);
  int lane = threadIdx.x & 31, wave = threadIdx.x >> 5;
  int half = lane >> 4, l16 = lane & 15;
#pragma unroll
  for (int t = 0; t < 4; ++t) {
    int c = bn * 64 + t * 16 + l16;
    if (c < 512) {
#pragma unroll
      for (int i = 0; i < 8; ++i) {
        int row = bm * 128 + wave * 16 + i + 8 * half;
        int b = row >> 11, n = row & 2047;
        float v = acc[t][i];
        if (c < 256) {        // Qr, fold 1/sqrt(RANK)=0.25
          int h = c >> 4, r = c & 15;
          Qr[(((size_t)(b * 16 + h) * 2048) + n) * 16 + r] = f2bf((v + bQr[c]) * 0.25f);
        } else {              // Kr, rank-padded to 32
          int c2 = c - 256, h = c2 >> 4, r = c2 & 15;
          size_t base = (((size_t)(b * 16 + h) * 2048) + n) * 32;
          Krp[base + r]      = f2bf(v + bKr[c2]);
          Krp[base + 16 + r] = 0;
        }
      }
    } else {                  // V -> transposed (b,h,v,n), one 16B store
      int cv = c - 512, h = cv >> 6, vc = cv & 63;
      int rowb = bm * 128 + wave * 16 + 8 * half;
      int b = rowb >> 11, nb = rowb & 2047;
      float bv_ = bV[vc];
      uint4 q;
      q.x = pk2(acc[t][0] + bv_, acc[t][1] + bv_);
      q.y = pk2(acc[t][2] + bv_, acc[t][3] + bv_);
      q.z = pk2(acc[t][4] + bv_, acc[t][5] + bv_);
      q.w = pk2(acc[t][6] + bv_, acc[t][7] + bv_);
      *(uint4*)(VT + (((size_t)(b * 16 + h) * 64 + vc) * 2048) + nb) = q;
    }
  }
}

// ---------------------------------------------------------------- kernel 4
// Flash attention in rank-16 space. 8 waves, 128 q rows, one (b,h) per block.
// Double-buffered async staging of Krp (32x32) and VT (64x32) chunks.
__global__ __launch_bounds__(256) void k_attn(
    const unsigned short* __restrict__ Qr, const unsigned short* __restrict__ Krp,
    const unsigned short* __restrict__ VT, unsigned short* __restrict__ attn) {
  __shared__ __attribute__((aligned(16))) unsigned short sKr[2 * 32 * 40];
  __shared__ __attribute__((aligned(16))) unsigned short sVT[2 * 64 * 40];
  __shared__ __attribute__((aligned(16))) unsigned short sP[8 * 16 * 32];
  int qblk = blockIdx.x & 15;      // 2048/128
  int bh   = blockIdx.x >> 4;      // b*16+h
  int tid = threadIdx.x, lane = tid & 31, wave = tid >> 5;
  int half = lane >> 4, l16 = lane & 15;
  const unsigned short* QrB = Qr  + (size_t)bh * 2048 * 16;
  const unsigned short* KrB = Krp + (size_t)bh * 2048 * 32;
  const unsigned short* VTB = VT  + (size_t)bh * 64 * 2048;

  // Qr A-fragment (rank padded 16 -> 32 with zeros in registers)
  FragAB aq;
  {
    int q = qblk * 128 + wave * 16 + l16;
    const unsigned int* qp = (const unsigned int*)(QrB + (size_t)q * 16);
#pragma unroll
    for (int j = 0; j < 4; ++j) { aq.u[j] = half ? 0u : qp[j]; aq.u[j + 4] = half ? 0u : qp[j + 4]; }
  }

  f32x8 o0 = zero8(), o1 = zero8(), o2 = zero8(), o3 = zero8();
  float mrow[8], lrow[8], srow[8];
#pragma unroll
  for (int i = 0; i < 8; ++i) { mrow[i] = -1e30f; lrow[i] = 0.f; }

  auto issueChunk = [&](int mb, int buf) {
    unsigned short* dK = sKr + buf * (32 * 40);
    unsigned short* dV = sVT + buf * (64 * 40);
    if (lane < 16) {                       // Kr chunk: 128 B128 xfers, EXEC=half
      int idx = wave * 16 + lane;          // 0..127
      int mm = idx >> 2, k = (idx & 3) * 8;
      cp_b128(KrB + (size_t)(mb + mm) * 32 + k, dK + mm * 40 + k);
    }
    {                                      // V^T chunk: 256 B128 xfers
      int vc = tid >> 2, ms = (tid & 3) * 8;
      cp_b128(VTB + (size_t)vc * 2048 + mb + ms, dV + vc * 40 + ms);
    }
  };

  issueChunk(0, 0);
  const int nIter = SEQ / 32;
  for (int it = 0; it < nIter; ++it) {
    int mb = it * 32, buf = it & 1;
    __syncthreads();                       // prev compute done; other buf free
    bool more = (it + 1) < nIter;
    if (more) issueChunk(mb + 32, buf ^ 1);
    if (more) wait_async<2>(); else wait_async<0>();
    __syncthreads();                       // current chunk visible
    const unsigned short* dK = sKr + buf * (32 * 40);
    const unsigned short* dV = sVT + buf * (64 * 40);

    // S(16x32) = Qr x Kr^T : two WMMAs
    f32x8 s0 = zero8(), s1 = zero8();
    {
      FragAB bk;
#pragma unroll
      for (int j = 0; j < 8; ++j)
        bk.u[j] = *(const unsigned int*)(dK + l16 * 40 + half * 16 + 2 * j);
      s0 = __builtin_amdgcn_wmma_f32_16x16x32_bf16(false, aq.v, false, bk.v,
                                                   (short)0, s0, false, false);
#pragma unroll
      for (int j = 0; j < 8; ++j)
        bk.u[j] = *(const unsigned int*)(dK + (16 + l16) * 40 + half * 16 + 2 * j);
      s1 = __builtin_amdgcn_wmma_f32_16x16x32_bf16(false, aq.v, false, bk.v,
                                                   (short)0, s1, false, false);
    }

    // streaming softmax over the 32-column chunk
    unsigned short* myP = sP + wave * 512;
#pragma unroll
    for (int i = 0; i < 8; ++i) {
      float t = fmaxf(s0[i], s1[i]);
      t = fmaxf(t, __shfl_xor(t, 1, 32));
      t = fmaxf(t, __shfl_xor(t, 2, 32));
      t = fmaxf(t, __shfl_xor(t, 4, 32));
      t = fmaxf(t, __shfl_xor(t, 8, 32));      // row max within 16-lane half
      float mn = fmaxf(mrow[i], t);
      float sc = __expf(mrow[i] - mn);
      float p0 = __expf(s0[i] - mn);
      float p1 = __expf(s1[i] - mn);
      float rs = p0 + p1;
      rs += __shfl_xor(rs, 1, 32);
      rs += __shfl_xor(rs, 2, 32);
      rs += __shfl_xor(rs, 4, 32);
      rs += __shfl_xor(rs, 8, 32);
      lrow[i] = lrow[i] * sc + rs;
      mrow[i] = mn; srow[i] = sc;
      int r = i + 8 * half;                    // C-layout -> A-layout transpose
      myP[r * 32 + l16]      = f2bf(p0);
      myP[r * 32 + 16 + l16] = f2bf(p1);
    }
    __syncthreads();                           // P visible / ordered

    // O = O*scale + P(16x32) x V(32x64)
#pragma unroll
    for (int i = 0; i < 8; ++i) {
      o0[i] *= srow[i]; o1[i] *= srow[i]; o2[i] *= srow[i]; o3[i] *= srow[i];
    }
    FragAB ap;
#pragma unroll
    for (int j = 0; j < 4; ++j) {
      int k0 = half * 16 + 2 * j;
      ap.u[j]     = *(const unsigned int*)(myP + l16 * 32 + k0);
      ap.u[j + 4] = *(const unsigned int*)(myP + l16 * 32 + 8 + k0);
    }
#pragma unroll
    for (int t = 0; t < 4; ++t) {
      FragAB bv;
#pragma unroll
      for (int j = 0; j < 8; ++j)
        bv.u[j] = *(const unsigned int*)(dV + (t * 16 + l16) * 40 + half * 16 + 2 * j);
      f32x8* po = (t == 0) ? &o0 : (t == 1) ? &o1 : (t == 2) ? &o2 : &o3;
      *po = __builtin_amdgcn_wmma_f32_16x16x32_bf16(false, ap.v, false, bv.v,
                                                    (short)0, *po, false, false);
    }
  }

  // epilogue: attn[b, n, h*64 + v] = O / l
  int b = bh >> 4, h = bh & 15;
#pragma unroll
  for (int i = 0; i < 8; ++i) {
    float inv = 1.0f / lrow[i];
    int n = qblk * 128 + wave * 16 + i + 8 * half;
    size_t base = ((size_t)(b * 2048 + n)) * 1024 + h * 64;
    attn[base +      l16] = f2bf(o0[i] * inv);
    attn[base + 16 + l16] = f2bf(o1[i] * inv);
    attn[base + 32 + l16] = f2bf(o2[i] * inv);
    attn[base + 48 + l16] = f2bf(o3[i] * inv);
  }
}

// ---------------------------------------------------------------- kernel 5
__global__ __launch_bounds__(256) void k_gemm_out(
    const unsigned short* __restrict__ attn, const unsigned short* __restrict__ WOt,
    const float* __restrict__ bO, float* __restrict__ out) {
  __shared__ __attribute__((aligned(16))) unsigned short sA[2 * 128 * 40];
  __shared__ __attribute__((aligned(16))) unsigned short sB[2 * 64 * 40];
  int bm = blockIdx.x & 31;          // 4096/128
  int bn = blockIdx.x >> 5;          // 1024/64
  f32x8 acc[4];
  gemm_tile_128x64(attn, 1024, bm * 128, WOt, 1024, bn * 64, 1024, acc, sA, sB);
  int lane = threadIdx.x & 31, wave = threadIdx.x >> 5;
  int half = lane >> 4, l16 = lane & 15;
#pragma unroll
  for (int t = 0; t < 4; ++t) {
    int c = bn * 64 + t * 16 + l16;
#pragma unroll
    for (int i = 0; i < 8; ++i) {
      int row = bm * 128 + wave * 16 + i + 8 * half;
      out[(size_t)row * 1024 + c] = acc[t][i] + bO[c];
    }
  }
}

// ---------------------------------------------------------------- launcher
extern "C" void kernel_launch(void* const* d_in, const int* in_sizes, int n_in,
                              void* d_out, int out_size, void* d_ws, size_t ws_size,
                              hipStream_t stream) {
  (void)in_sizes; (void)n_in; (void)out_size; (void)ws_size;
  const float* x   = (const float*)d_in[0];
  const float* WQ  = (const float*)d_in[1];
  const float* bQ  = (const float*)d_in[2];
  const float* WK  = (const float*)d_in[3];
  const float* bK  = (const float*)d_in[4];
  const float* WV  = (const float*)d_in[5];
  const float* bV  = (const float*)d_in[6];
  const float* WO  = (const float*)d_in[7];
  const float* bO  = (const float*)d_in[8];
  const float* WQr = (const float*)d_in[9];
  const float* WKr = (const float*)d_in[10];
  float* out = (float*)d_out;

  char* ws = (char*)d_ws;
  unsigned short* xb   = (unsigned short*)(ws + 0);          //  8 MB
  unsigned short* WfT  = (unsigned short*)(ws + 8388608);    //  3 MB
  unsigned short* WOt  = (unsigned short*)(ws + 11534336);   //  2 MB
  unsigned short* Qr   = (unsigned short*)(ws + 13631488);   //  2 MB
  unsigned short* Krp  = (unsigned short*)(ws + 15728640);   //  4 MB (rank-padded)
  unsigned short* VT   = (unsigned short*)(ws + 19922944);   //  8 MB (transposed)
  float*          bQrw = (float*)(ws + 28311552);
  float*          bKrw = (float*)(ws + 28312576);
  unsigned short* attn = xb;  // xb dead after k_gemm_qkv (stream-ordered reuse)

  k_convert_x<<<(ROWS * D_MODEL) / 256, 256, 0, stream>>>(x, xb);
  k_fuse_weights<<<((1536 + 1024) * 1024) / 256, 256, 0, stream>>>(
      WQ, WK, WV, WO, WQr, WKr, bQ, bK, WfT, WOt, bQrw, bKrw);
  k_gemm_qkv<<<32 * 24, 256, 0, stream>>>(xb, WfT, bQrw, bKrw, bV, Qr, Krp, VT);
  k_attn<<<BATCH * N_HEADS * (SEQ / 128), 256, 0, stream>>>(Qr, Krp, VT, attn);
  k_gemm_out<<<32 * 16, 256, 0, stream>>>(attn, WOt, bO, out);
}
